// MambaFNO_AvgPool_52982716563643
// MI455X (gfx1250) — matrix operand order repaired and checked by reading, not compiled, not run on hardware
//
#include <hip/hip_runtime.h>

// ------------------------------------------------------------------
// MambaFNO forward for MI455X (gfx1250, wave32, WMMA + TDM).
// Activations token-major (HW x C). Big per-token GEMMs run on
// v_wmma_f32_16x16x32_f16 (f32 accumulate) with TDM-staged LDS tiles
// converted once into fragment-ordered f16 so the compute loop does
// contiguous b128 LDS fragment loads (no scalar ds_load_u16 chains).
// ------------------------------------------------------------------

#define HW     16384
#define HH     128
#define WWID   128
#define C64    64
#define DIN    128
#define XZDIM  256
#define NSTATE 8
#define DTRNK  4
#define KCONV  4
#define MODES  16
#define NCLS   10
#define GRP    4
#define XPD    20      /* DTR + 2*NS */
#define CHUNK  128
#define NCHUNK 128
#define KMAX   128     /* max GEMM K in this model */

typedef __attribute__((ext_vector_type(16))) _Float16 v16h;
typedef __attribute__((ext_vector_type(4)))  _Float16 v4h;
typedef __attribute__((ext_vector_type(8)))  float    v8f;
typedef __attribute__((ext_vector_type(4)))  unsigned int v4u;
typedef __attribute__((ext_vector_type(4)))  int      v4i;
typedef __attribute__((ext_vector_type(8)))  int      v8i;

__device__ __forceinline__ float siluf(float v) { return v / (1.0f + __expf(-v)); }

#if __has_builtin(__builtin_amdgcn_tensor_load_to_lds)
#define HAVE_TDM 1
#else
#define HAVE_TDM 0
#endif

#if HAVE_TDM
// 2-D TDM tile load: tile_rows x rowlen f32 elements, row stride = stride
// elements, from gaddr into LDS at lds_off. D# packing per CDNA5 ISA 8.3/8.4.
__device__ __forceinline__ void tdm_load_2d(unsigned lds_off, const void* gaddr,
                                            unsigned rowlen, unsigned nrows_tensor,
                                            unsigned tile_rows, unsigned stride)
{
    unsigned long long ga = (unsigned long long)(uintptr_t)gaddr;
    v4u g0;
    g0[0] = 1u;                                        // count=1 (user D#)
    g0[1] = lds_off;                                   // lds_addr
    g0[2] = (unsigned)(ga & 0xffffffffull);            // global_addr[31:0]
    g0[3] = (unsigned)((ga >> 32) & 0x1ffffffull) | (2u << 30); // addr[56:32] | type=2
    v8i g1 = {0, 0, 0, 0, 0, 0, 0, 0};
    g1[0] = (2 << 16);                                 // data_size = 4B, no iterate/pad
    g1[1] = (int)((rowlen & 0xffffu) << 16);           // tensor_dim0[15:0]
    g1[2] = (int)((rowlen >> 16) & 0xffffu) |          // tensor_dim0[31:16]
            (int)((nrows_tensor & 0xffffu) << 16);     // tensor_dim1[15:0]
    g1[3] = (int)((nrows_tensor >> 16) & 0xffffu) |    // tensor_dim1[31:16]
            (int)((rowlen & 0xffffu) << 16);           // tile_dim0
    g1[4] = (int)(tile_rows & 0xffffu);                // tile_dim1 (tile_dim2 = 0)
    g1[5] = (int)stride;                               // tensor_dim0_stride[31:0]
    v4i z4 = {0, 0, 0, 0};
#if defined(__clang_major__) && (__clang_major__ >= 23)
    v8i z8 = {0, 0, 0, 0, 0, 0, 0, 0};
    __builtin_amdgcn_tensor_load_to_lds(g0, g1, z4, z4, z8, 0);
#else
    __builtin_amdgcn_tensor_load_to_lds(g0, g1, z4, z4, 0);
#endif
}
#endif

// ------------------------------------------------------------------
// D[M x N] = act(A[M x K] @ Wt[N x K]^T + bias)
// grid = (M/16, ceil(N/64)), block = 32 (one wave); 16x64 tile per wave.
// Phase 1: TDM (or vectorized manual) stage of A(16xK) and B(64xK) f32.
// Phase 2: one conversion pass into fragment-ordered f16 LDS.
// Phase 3: barrier-free K loop: v16h fragment loads + 4x wmma per step.
// ------------------------------------------------------------------
__global__ void k_gemm_wmma(const float* __restrict__ A,
                            const float* __restrict__ Wt,
                            const float* __restrict__ bias,
                            float* __restrict__ D,
                            int M, int N, int K, int act)
{
    __shared__ float    sAst[16 * KMAX];          // 8 KB
    __shared__ float    sBst[64 * KMAX];          // 32 KB
    __shared__ _Float16 fA[16 * KMAX];            // frag-order, 4 KB
    __shared__ _Float16 fB[64 * KMAX];            // frag-order, 16 KB
    const int lane = threadIdx.x;
    const int m0   = blockIdx.x * 16;
    const int n0   = blockIdx.y * 64;

    // ---- phase 1: stage f32 stripes into LDS -------------------------------
#if HAVE_TDM
    {
        unsigned la = (unsigned)(uintptr_t)(&sAst[0]);
        unsigned lb = (unsigned)(uintptr_t)(&sBst[0]);
        tdm_load_2d(la, A + (size_t)m0 * K, (unsigned)K, 16u, 16u, (unsigned)K);
        unsigned brows = (N > n0) ? (unsigned)(N - n0) : 0u;
        tdm_load_2d(lb, Wt + (size_t)n0 * K, (unsigned)K, brows, 64u, (unsigned)K);
#if __has_builtin(__builtin_amdgcn_s_wait_tensorcnt)
        __builtin_amdgcn_s_wait_tensorcnt(0);
#endif
    }
#else
    for (int q = lane; q < (16 * K) / 4; q += 32) {
        int r = (q * 4) / K, kk = (q * 4) % K;
        *(float4*)&sAst[r * K + kk] = *(const float4*)&A[(size_t)(m0 + r) * K + kk];
    }
    for (int q = lane; q < (64 * K) / 4; q += 32) {
        int n = (q * 4) / K, kk = (q * 4) % K;
        float4 v = make_float4(0.f, 0.f, 0.f, 0.f);
        if (n0 + n < N) v = *(const float4*)&Wt[(size_t)(n0 + n) * K + kk];
        *(float4*)&sBst[n * K + kk] = v;
    }
#endif
    __syncthreads();

    // ---- phase 2: convert to fragment-ordered f16 --------------------------
    // A frag (per 32-k step ks): element e of lane (hi*16+row) holds
    //   k = ks*32 + hi*8 + (e&7) + 16*(e>>3)  ->  inverse used below.
    for (int q = lane; q < (16 * K) / 4; q += 32) {
        int r  = (q * 4) / K, kk = (q * 4) % K;
        int ks = kk >> 5, kl = kk & 31;
        int hi = (kl >> 3) & 1;
        int e  = (kl & 7) + ((kl >> 4) << 3);
        float4 v = *(const float4*)&sAst[r * K + kk];
        v4h p; p[0] = (_Float16)v.x; p[1] = (_Float16)v.y;
               p[2] = (_Float16)v.z; p[3] = (_Float16)v.w;
        *(v4h*)&fA[(ks * 32 + hi * 16 + r) * 16 + e] = p;
    }
    // B frag: element e of lane (hi*16+nc), tile j = n>>4: k = ks*32+hi*8+(e&7)+16*(e>>3).
    for (int q = lane; q < (64 * K) / 4; q += 32) {
        int n  = (q * 4) / K, kk = (q * 4) % K;
        int ks = kk >> 5, kl = kk & 31;
        int hi = (kl >> 3) & 1, e = (kl & 7) + ((kl >> 4) << 3);
        int j  = n >> 4,  nc = n & 15;
        float4 v = *(const float4*)&sBst[n * K + kk];
        v4h p; p[0] = (_Float16)v.x; p[1] = (_Float16)v.y;
               p[2] = (_Float16)v.z; p[3] = (_Float16)v.w;
        *(v4h*)&fB[(((ks * 4 + j) * 32) + hi * 16 + nc) * 16 + e] = p;
    }
    __syncthreads();

    // ---- phase 3: barrier-free wmma loop -----------------------------------
    v8f acc0 = {}, acc1 = {}, acc2 = {}, acc3 = {};
    const int nks = K >> 5;
    for (int ks = 0; ks < nks; ++ks) {
        v16h af = *(const v16h*)&fA[(ks * 32 + lane) * 16];
        v16h b0 = *(const v16h*)&fB[(((ks * 4 + 0) * 32) + lane) * 16];
        v16h b1 = *(const v16h*)&fB[(((ks * 4 + 1) * 32) + lane) * 16];
        v16h b2 = *(const v16h*)&fB[(((ks * 4 + 2) * 32) + lane) * 16];
        v16h b3 = *(const v16h*)&fB[(((ks * 4 + 3) * 32) + lane) * 16];
        acc0 = __builtin_amdgcn_wmma_f32_16x16x32_f16(false, af, false, b0, (short)0, acc0, false, false);
        acc1 = __builtin_amdgcn_wmma_f32_16x16x32_f16(false, af, false, b1, (short)0, acc1, false, false);
        acc2 = __builtin_amdgcn_wmma_f32_16x16x32_f16(false, af, false, b2, (short)0, acc2, false, false);
        acc3 = __builtin_amdgcn_wmma_f32_16x16x32_f16(false, af, false, b3, (short)0, acc3, false, false);
    }

    // ---- store: C/D layout: vgpr r -> M = r + 8*(lane>=16), N = lane&15 ----
    const int row = lane & 15;
    const int hi  = lane >> 4;
#pragma unroll
    for (int r = 0; r < 8; ++r) {
        int mr = m0 + r + hi * 8;
        v8f* accs[4] = {&acc0, &acc1, &acc2, &acc3};
#pragma unroll
        for (int j = 0; j < 4; ++j) {
            int nc = n0 + j * 16 + row;
            if (nc < N) {
                float v = (*accs[j])[r];
                if (bias) v += bias[nc];
                if (act)  v = siluf(v);
                D[(size_t)mr * N + nc] = v;
            }
        }
    }
}

// ---------------- patch-embed 1x1 conv (Cin=3), writes token-major -----------
__global__ void k_pe(const float* __restrict__ x0, const float* __restrict__ w,
                     const float* __restrict__ b, float* __restrict__ y)
{
    int p = blockIdx.x * blockDim.x + threadIdx.x;
    if (p >= HW) return;
    float v0 = x0[p], v1 = x0[HW + p], v2 = x0[2 * HW + p];
    for (int o = 0; o < C64; ++o)
        y[p * C64 + o] = w[o * 3] * v0 + w[o * 3 + 1] * v1 + w[o * 3 + 2] * v2 + b[o];
}

// ---------------- GroupNorm (token-major x: HW x Cc) -------------------------
__global__ void k_gn_stats(const float* __restrict__ x, float* __restrict__ stats,
                           int Cc, int groups)
{
    int g  = blockIdx.x;
    int cg = Cc / groups;
    __shared__ float ssum[256];
    __shared__ float ssq[256];
    float s = 0.0f, q = 0.0f;
    for (int i = threadIdx.x; i < HW * cg; i += blockDim.x) {
        int p = i / cg;
        int c = g * cg + (i % cg);
        float v = x[p * Cc + c];
        s += v; q += v * v;
    }
    ssum[threadIdx.x] = s; ssq[threadIdx.x] = q;
    __syncthreads();
    for (int st = 128; st > 0; st >>= 1) {
        if ((int)threadIdx.x < st) {
            ssum[threadIdx.x] += ssum[threadIdx.x + st];
            ssq[threadIdx.x]  += ssq[threadIdx.x + st];
        }
        __syncthreads();
    }
    if (threadIdx.x == 0) {
        float n   = (float)(HW * cg);
        float mu  = ssum[0] / n;
        float var = ssq[0] / n - mu * mu;
        stats[2 * g]     = mu;
        stats[2 * g + 1] = rsqrtf(var + 1e-5f);
    }
}

__global__ void k_gn_apply(const float* __restrict__ x, float* __restrict__ y,
                           const float* __restrict__ stats,
                           const float* __restrict__ gamma, const float* __restrict__ beta,
                           int Cc, int groups, int act)
{
    int i = blockIdx.x * blockDim.x + threadIdx.x;
    if (i >= HW * Cc) return;
    int c = i % Cc;
    int g = c / (Cc / groups);
    float v = (x[i] - stats[2 * g]) * stats[2 * g + 1] * gamma[c] + beta[c];
    if (act) v = siluf(v);
    y[i] = v;
}

// ---------------- causal / anticausal depthwise conv + SiLU ------------------
__global__ void k_dwconv_silu(const float* __restrict__ xz, const float* __restrict__ w,
                              const float* __restrict__ b, float* __restrict__ u, int rev)
{
    int i = blockIdx.x * blockDim.x + threadIdx.x;
    if (i >= HW * DIN) return;
    int l = i / DIN, d = i % DIN;
    float acc = b[d];
#pragma unroll
    for (int j = 0; j < KCONV; ++j) {
        int lj = rev ? (l + (KCONV - 1) - j) : (l - (KCONV - 1) + j);
        if (lj >= 0 && lj < HW) acc += w[d * KCONV + j] * xz[lj * XZDIM + d];
    }
    u[i] = siluf(acc);
}

// ---------------- dt = softplus(dbl[:, :4] @ W_dt^T + bias) ------------------
__global__ void k_dt(const float* __restrict__ dbl, const float* __restrict__ wdt,
                     const float* __restrict__ dtb, float* __restrict__ dt)
{
    int i = blockIdx.x * blockDim.x + threadIdx.x;
    if (i >= HW * DIN) return;
    int l = i / DIN, d = i % DIN;
    float a = dtb[d];
#pragma unroll
    for (int r = 0; r < DTRNK; ++r) a += dbl[l * XPD + r] * wdt[d * DTRNK + r];
    dt[i] = (a > 20.0f) ? a : log1pf(__expf(a));
}

// ---------------- chunked associative scan -----------------------------------
__global__ void k_scan_pass1(const float* __restrict__ dt, const float* __restrict__ u,
                             const float* __restrict__ dbl, const float* __restrict__ alog,
                             float* __restrict__ Acar, float* __restrict__ Bcar, int rev)
{
    int ch = blockIdx.x;
    int d  = threadIdx.x;            // 0..127
    float An[NSTATE], Ar[NSTATE], Br[NSTATE];
#pragma unroll
    for (int n = 0; n < NSTATE; ++n) {
        An[n] = -__expf(alog[d * NSTATE + n]);
        Ar[n] = 1.0f; Br[n] = 0.0f;
    }
    for (int s = 0; s < CHUNK; ++s) {
        int sidx = ch * CHUNK + s;
        int l    = rev ? (HW - 1 - sidx) : sidx;
        if (s + 1 < CHUNK) {
            int ln = rev ? (l - 1) : (l + 1);
            __builtin_prefetch(&dt[ln * DIN + d], 0, 1);
            __builtin_prefetch(&u[ln * DIN + d], 0, 1);
        }
        float dtv = dt[l * DIN + d];
        float du  = dtv * u[l * DIN + d];
#pragma unroll
        for (int n = 0; n < NSTATE; ++n) {
            float a  = __expf(dtv * An[n]);
            float bb = du * dbl[l * XPD + DTRNK + n];
            Ar[n] *= a;
            Br[n]  = a * Br[n] + bb;
        }
    }
#pragma unroll
    for (int n = 0; n < NSTATE; ++n) {
        Acar[(ch * DIN + d) * NSTATE + n] = Ar[n];
        Bcar[(ch * DIN + d) * NSTATE + n] = Br[n];
    }
}

__global__ void k_scan_pass2(const float* __restrict__ Acar, const float* __restrict__ Bcar,
                             float* __restrict__ Hin)
{
    int i = blockIdx.x * blockDim.x + threadIdx.x;   // d*NS + n
    if (i >= DIN * NSTATE) return;
    float h = 0.0f;
    for (int ch = 0; ch < NCHUNK; ++ch) {
        Hin[ch * DIN * NSTATE + i] = h;
        h = Acar[ch * DIN * NSTATE + i] * h + Bcar[ch * DIN * NSTATE + i];
    }
}

__global__ void k_scan_pass3(const float* __restrict__ dt, const float* __restrict__ u,
                             const float* __restrict__ dbl, const float* __restrict__ alog,
                             const float* __restrict__ Dp, const float* __restrict__ Hin,
                             const float* __restrict__ xz, float* __restrict__ ybr, int rev)
{
    int ch = blockIdx.x;
    int d  = threadIdx.x;
    float An[NSTATE], h[NSTATE];
#pragma unroll
    for (int n = 0; n < NSTATE; ++n) {
        An[n] = -__expf(alog[d * NSTATE + n]);
        h[n]  = Hin[(ch * DIN + d) * NSTATE + n];
    }
    float Dv = Dp[d];
    for (int s = 0; s < CHUNK; ++s) {
        int sidx = ch * CHUNK + s;
        int l    = rev ? (HW - 1 - sidx) : sidx;
        if (s + 1 < CHUNK) {
            int ln = rev ? (l - 1) : (l + 1);
            __builtin_prefetch(&dt[ln * DIN + d], 0, 1);
            __builtin_prefetch(&u[ln * DIN + d], 0, 1);
        }
        float dtv = dt[l * DIN + d];
        float uv  = u[l * DIN + d];
        float du  = dtv * uv;
        float y   = 0.0f;
#pragma unroll
        for (int n = 0; n < NSTATE; ++n) {
            float a = __expf(dtv * An[n]);
            h[n] = a * h[n] + du * dbl[l * XPD + DTRNK + n];
            y   += h[n] * dbl[l * XPD + DTRNK + NSTATE + n];
        }
        y += Dv * uv;
        float z = xz[l * XZDIM + DIN + d];
        ybr[l * DIN + d] = y * siluf(z);
    }
}

// ---------------- 2-way attention gate + residual ----------------------------
__global__ void k_att_combine(const float* __restrict__ t, const float* __restrict__ watt,
                              const float* __restrict__ batt,
                              const float* __restrict__ yf, const float* __restrict__ yb,
                              const float* __restrict__ xin, float* __restrict__ xout)
{
    int l = blockIdx.x * blockDim.x + threadIdx.x;
    if (l >= HW) return;
    float a0 = batt[0], a1 = batt[1];
    for (int c = 0; c < C64; ++c) {
        float tv = t[l * C64 + c];
        a0 += watt[c] * tv;
        a1 += watt[C64 + c] * tv;
    }
    float m  = fmaxf(a0, a1);
    float e0 = __expf(a0 - m), e1 = __expf(a1 - m);
    float inv = 1.0f / (e0 + e1);
    e0 *= inv; e1 *= inv;
    for (int c = 0; c < C64; ++c) {
        int i = l * C64 + c;
        xout[i] = xin[i] + e0 * yf[i] + e1 * yb[i];
    }
}

// ---------------- separable spectral layer -----------------------------------
__global__ void k_twiddle(float* __restrict__ cosT, float* __restrict__ sinT)
{
    int i = blockIdx.x * blockDim.x + threadIdx.x;  // k*128 + x
    if (i >= 128 * 128) return;
    int k = i >> 7, x = i & 127;
    float ang = 6.28318530717958647692f * (float)((k * x) & 127) / 128.0f;
    cosT[i] = __cosf(ang);
    sinT[i] = __sinf(ang);
}

__global__ void k_dft_w(const float* __restrict__ x, const float* __restrict__ cosT,
                        const float* __restrict__ sinT, float* __restrict__ specA)
{
    int i = blockIdx.x * blockDim.x + threadIdx.x;  // c*H*M + h*M + ky
    if (i >= C64 * HH * MODES) return;
    int ky = i % MODES;
    int h  = (i / MODES) % HH;
    int c  = i / (MODES * HH);
    const float* ct = cosT + ky * 128;
    const float* st = sinT + ky * 128;
    float re = 0.0f, im = 0.0f;
    for (int w = 0; w < WWID; ++w) {
        float v = x[(h * WWID + w) * C64 + c];
        re += v * ct[w];
        im -= v * st[w];
    }
    specA[2 * i] = re; specA[2 * i + 1] = im;
}

__global__ void k_dft_h(const float* __restrict__ specA, const float* __restrict__ cosT,
                        const float* __restrict__ sinT, float* __restrict__ xfm)
{
    int i = blockIdx.x * blockDim.x + threadIdx.x;  // c*32*M + kxi*M + ky
    if (i >= C64 * 32 * MODES) return;
    int ky  = i % MODES;
    int kxi = (i / MODES) % 32;
    int c   = i / (MODES * 32);
    int kx  = (kxi < MODES) ? kxi : (96 + kxi);
    float re = 0.0f, im = 0.0f;
    for (int h = 0; h < HH; ++h) {
        int j = (c * HH + h) * MODES + ky;
        float ar = specA[2 * j], ai = specA[2 * j + 1];
        float ctv = cosT[kx * 128 + h], stv = sinT[kx * 128 + h];
        re += ar * ctv + ai * stv;       // (ar+i ai)(cos - i sin)
        im += ai * ctv - ar * stv;
    }
    xfm[2 * i] = re; xfm[2 * i + 1] = im;
}

__global__ void k_spec_mix(const float* __restrict__ xfm, const float* __restrict__ wspec,
                           float* __restrict__ ofm)
{
    int i = blockIdx.x * blockDim.x + threadIdx.x;  // o*32*M + kxi*M + ky
    if (i >= C64 * 32 * MODES) return;
    int ky  = i % MODES;
    int kxi = (i / MODES) % 32;
    int o   = i / (MODES * 32);
    int s   = (kxi < MODES) ? 0 : 1;
    int kx  = kxi & (MODES - 1);
    float re = 0.0f, im = 0.0f;
    for (int ci = 0; ci < C64; ++ci) {
        const float* wp = wspec + ((((size_t)(s * C64 + ci) * C64 + o) * MODES + kx) * MODES + ky) * 2;
        int j = (ci * 32 + kxi) * MODES + ky;
        float xr = xfm[2 * j], xi = xfm[2 * j + 1];
        float wr = wp[0], wi = wp[1];
        re += xr * wr - xi * wi;
        im += xr * wi + xi * wr;
    }
    ofm[2 * i] = re; ofm[2 * i + 1] = im;
}

__global__ void k_idft_h(const float* __restrict__ ofm, const float* __restrict__ cosT,
                         const float* __restrict__ sinT, float* __restrict__ gsp)
{
    int i = blockIdx.x * blockDim.x + threadIdx.x;  // c*H*M + h*M + ky
    if (i >= C64 * HH * MODES) return;
    int ky = i % MODES;
    int h  = (i / MODES) % HH;
    int c  = i / (MODES * HH);
    float re = 0.0f, im = 0.0f;
    for (int kxi = 0; kxi < 32; ++kxi) {
        int kx = (kxi < MODES) ? kxi : (96 + kxi);
        int j  = (c * 32 + kxi) * MODES + ky;
        float orv = ofm[2 * j], oiv = ofm[2 * j + 1];
        float ctv = cosT[kx * 128 + h], stv = sinT[kx * 128 + h];
        re += orv * ctv - oiv * stv;     // (or+i oi)(cos + i sin)
        im += orv * stv + oiv * ctv;
    }
    gsp[2 * i] = re; gsp[2 * i + 1] = im;
}

__global__ void k_idft_w_combine(const float* __restrict__ gsp, const float* __restrict__ cosT,
                                 const float* __restrict__ sinT, const float* __restrict__ byp,
                                 float* __restrict__ xout)
{
    int i = blockIdx.x * blockDim.x + threadIdx.x;  // p*C + c
    if (i >= HW * C64) return;
    int c = i % C64;
    int p = i / C64;
    int w = p % WWID;
    int h = p / WWID;
    float acc = 0.0f;
#pragma unroll
    for (int ky = 0; ky < MODES; ++ky) {
        int j = (c * HH + h) * MODES + ky;
        float gr = gsp[2 * j], gi = gsp[2 * j + 1];
        float term = gr * cosT[ky * 128 + w] - gi * sinT[ky * 128 + w];
        acc += (ky == 0) ? term : 2.0f * term;
    }
    float v = acc * (1.0f / 16384.0f) + byp[i];
    xout[i] = siluf(v);
}

// ---------------- classifier head (Cout=10, channel-major output) ------------
__global__ void k_head2(const float* __restrict__ hdn, const float* __restrict__ w2,
                        const float* __restrict__ b2, float* __restrict__ out)
{
    int p = blockIdx.x * blockDim.x + threadIdx.x;
    if (p >= HW) return;
    for (int o = 0; o < NCLS; ++o) {
        float a = b2[o];
        for (int c = 0; c < DIN; ++c) a += hdn[p * DIN + c] * w2[o * DIN + c];
        out[o * HW + p] = a;
    }
}

// ------------------------------------------------------------------
// Input flattening (depth-first, dict insertion order):
//  0 x | 1 pe_w 2 pe_b 3 pe_g 4 pe_be
//  mamba block (22): +0 gn_g +1 gn_b +2..10 fwd{W_in,conv_w,conv_b,W_xproj,
//    W_dt,dt_bias,A_log,Dp,W_out} +11..19 bwd{...} +20 W_att +21 b_att
//  fno block (3): +0 w_spec +1 w_byp +2 b_byp
//  m0@5  f0@27  m1@30  f1@52  m2@55
//  77 h1_w 78 h1_b 79 h_g 80 h_be 81 h2_w 82 h2_b
// ------------------------------------------------------------------
extern "C" void kernel_launch(void* const* d_in, const int* in_sizes, int n_in,
                              void* d_out, int out_size, void* d_ws, size_t ws_size,
                              hipStream_t stream)
{
    (void)in_sizes; (void)out_size; (void)ws_size;
    const float* in[96];
    for (int i = 0; i < n_in && i < 96; ++i) in[i] = (const float*)d_in[i];

    char* wsb = (char*)d_ws;
    size_t off = 0;
    auto alloc = [&](size_t nfloats) -> float* {
        float* p = (float*)(wsb + off);
        off += ((nfloats * sizeof(float) + 255) / 256) * 256;
        return p;
    };

    float* xA    = alloc((size_t)HW * C64);
    float* xB    = alloc((size_t)HW * C64);
    float* tbuf  = alloc((size_t)HW * C64);
    float* xz    = alloc((size_t)HW * XZDIM);
    float* ubuf  = alloc((size_t)HW * DIN);
    float* dblb  = alloc((size_t)HW * XPD);
    float* dtb   = alloc((size_t)HW * DIN);
    float* ybr   = alloc((size_t)HW * DIN);
    float* yfb   = alloc((size_t)HW * C64);
    float* ybb   = alloc((size_t)HW * C64);
    float* Acar  = alloc((size_t)NCHUNK * DIN * NSTATE);
    float* Bcar  = alloc((size_t)NCHUNK * DIN * NSTATE);
    float* Hin   = alloc((size_t)NCHUNK * DIN * NSTATE);
    float* stats = alloc(16);
    float* cosT  = alloc(128 * 128);
    float* sinT  = alloc(128 * 128);
    float* specA = alloc((size_t)C64 * HH * MODES * 2);
    float* xfm   = alloc((size_t)C64 * 32 * MODES * 2);
    float* ofm   = alloc((size_t)C64 * 32 * MODES * 2);
    float* gsp   = alloc((size_t)C64 * HH * MODES * 2);
    float* byp   = alloc((size_t)HW * C64);
    float* hdn   = alloc((size_t)HW * DIN);
    float* hdn2  = alloc((size_t)HW * DIN);

    const int TB = 256;
    auto gemm = [&](const float* A, const float* Wt, const float* bias, float* D,
                    int M, int N, int K, int act) {
        dim3 grid(M / 16, (N + 63) / 64);
        k_gemm_wmma<<<grid, dim3(32), 0, stream>>>(A, Wt, bias, D, M, N, K, act);
    };

    auto run_branch = [&](const float* const* bp, float* yout, int rev) {
        gemm(tbuf, bp[0], nullptr, xz, HW, XZDIM, C64, 0);                       // W_in
        k_dwconv_silu<<<(HW * DIN + TB - 1) / TB, TB, 0, stream>>>(xz, bp[1], bp[2], ubuf, rev);
        gemm(ubuf, bp[3], nullptr, dblb, HW, XPD, DIN, 0);                       // W_xproj
        k_dt<<<(HW * DIN + TB - 1) / TB, TB, 0, stream>>>(dblb, bp[4], bp[5], dtb);
        k_scan_pass1<<<NCHUNK, DIN, 0, stream>>>(dtb, ubuf, dblb, bp[6], Acar, Bcar, rev);
        k_scan_pass2<<<4, 256, 0, stream>>>(Acar, Bcar, Hin);
        k_scan_pass3<<<NCHUNK, DIN, 0, stream>>>(dtb, ubuf, dblb, bp[6], bp[7], Hin, xz, ybr, rev);
        gemm(ybr, bp[8], nullptr, yout, HW, C64, DIN, 0);                        // W_out
    };

    auto run_mamba = [&](int base, const float* xin, float* xout) {
        k_gn_stats<<<GRP, 256, 0, stream>>>(xin, stats, C64, GRP);
        k_gn_apply<<<(HW * C64 + TB - 1) / TB, TB, 0, stream>>>(
            xin, tbuf, stats, in[base + 0], in[base + 1], C64, GRP, 0);
        run_branch(&in[base + 2], yfb, 0);
        run_branch(&in[base + 11], ybb, 1);
        k_att_combine<<<(HW + TB - 1) / TB, TB, 0, stream>>>(
            tbuf, in[base + 20], in[base + 21], yfb, ybb, xin, xout);
    };

    auto run_fno = [&](int base, const float* xin, float* xout) {
        gemm(xin, in[base + 1], in[base + 2], byp, HW, C64, C64, 0);             // bypass
        k_dft_w<<<(C64 * HH * MODES + TB - 1) / TB, TB, 0, stream>>>(xin, cosT, sinT, specA);
        k_dft_h<<<(C64 * 32 * MODES + TB - 1) / TB, TB, 0, stream>>>(specA, cosT, sinT, xfm);
        k_spec_mix<<<(C64 * 32 * MODES + TB - 1) / TB, TB, 0, stream>>>(xfm, in[base + 0], ofm);
        k_idft_h<<<(C64 * HH * MODES + TB - 1) / TB, TB, 0, stream>>>(ofm, cosT, sinT, gsp);
        k_idft_w_combine<<<(HW * C64 + TB - 1) / TB, TB, 0, stream>>>(gsp, cosT, sinT, byp, xout);
    };

    // twiddle tables (recomputed each call: deterministic, no cached state)
    k_twiddle<<<(128 * 128 + TB - 1) / TB, TB, 0, stream>>>(cosT, sinT);

    // stem: conv1x1(3->64) -> GN -> SiLU
    k_pe<<<(HW + TB - 1) / TB, TB, 0, stream>>>(in[0], in[1], in[2], xA);
    k_gn_stats<<<GRP, 256, 0, stream>>>(xA, stats, C64, GRP);
    k_gn_apply<<<(HW * C64 + TB - 1) / TB, TB, 0, stream>>>(
        xA, xB, stats, in[3], in[4], C64, GRP, 1);

    run_mamba(5,  xB, xA);
    run_fno (27, xA, xB);
    run_mamba(30, xB, xA);
    run_fno (52, xA, xB);
    run_mamba(55, xB, xA);

    // head: conv1x1(64->128) -> GN -> SiLU -> conv1x1(128->10)
    gemm(xA, in[77], in[78], hdn, HW, DIN, C64, 0);
    k_gn_stats<<<GRP, 256, 0, stream>>>(hdn, stats, DIN, GRP);
    k_gn_apply<<<(HW * DIN + TB - 1) / TB, TB, 0, stream>>>(
        hdn, hdn2, stats, in[79], in[80], DIN, GRP, 1);
    k_head2<<<(HW + TB - 1) / TB, TB, 0, stream>>>(hdn2, in[81], in[82], (float*)d_out);
}
